// RoIAlignAvg_51805895525041
// MI455X (gfx1250) — matrix operand order repaired and checked by reading, not compiled, not run on hardware
//
#include <hip/hip_runtime.h>
#include <hip/hip_bf16.h>

// RoIAlign(avg, PH=PW=8, SR=2, scale=1/16) + 2x2 window average, for MI455X.
//
// Design (v6 = v4 + distinct B staging; window-branch experiment reverted):
//  * grid = K RoIs, block = 256 threads (8 wave32), 2 channel-tiles per wave.
//  * Per block: precompute 16 y-tap and 16 x-tap parameter sets into LDS.
//  * Each lane owns one channel; per 32-sample K-chunk it computes a 2x8
//    patch of bilinear samples == exactly its A-operand slice for
//    V_WMMA_F32_16X16X32_F16 (16-bit A 16x32 layout: lane = M = channel;
//    element e -> sy = 2*chunk + (e>>3), sx = 8*(lane>>4) + (e&7)).
//  * Pooling (mean over 2x2 samples then 2x2 bins == 1/16 * 4x4 sample
//    window) is the constant 256x49 matrix P (entries 0 or 1/16, exact in
//    f16), contracted on the matrix pipes with f32 accumulation.
//  * Paired-tap gathers: both x-taps fetched with one dword-aligned b64 at
//    xb = min(x0, W-2); lx' = (x0==W-1) ? 1 : lx is bit-exact at the clamp.
//  * v6 notes:
//      - WMMAs are UNCONDITIONAL on a single straight-line path: guarding
//        them with scalar chunk-window branches (v5) made LLVM insert
//        v_mov_b64 phi-copy chains for the accumulators that cost far more
//        than the skipped zero-B WMMAs.
//      - The four masked B operands are built into distinct variables before
//        the WMMA block, breaking the shared-staging WAR chain that forced
//        4 v_nops after every WMMA.
//  * Kept: B factorization (Bpat per nt, one compare + 8 dword selects per
//    mask; okc <=> (unsigned)(chunk - i) < 2, g-independent), non-temporal
//    output stores (keep the 192 MB L2 for the 249 MB feature map), rolled
//    K-chunk loop (<256 VGPRs, high occupancy), scalar batch base + 32-bit
//    lane offsets (saddr-form loads), global_prefetch of next chunk's rows.

typedef __attribute__((ext_vector_type(16))) _Float16 v16h;
typedef __attribute__((ext_vector_type(8)))  float    v8f;
typedef __attribute__((ext_vector_type(8)))  int      v8i;
typedef __attribute__((ext_vector_type(2)))  float    f2v;

#define SCALE_F 0.0625f
#define C_DIM   256
#define H_DIM   200
#define W_DIM   304
#define HW_DIM  (H_DIM * W_DIM)
#define NOUT    49      // 7*7 pooled outputs per channel

// Align-4 pair load: lets the backend emit one global_load_b64 per tap pair.
static __device__ __forceinline__ f2v ld_pair(const float* __restrict__ p) {
    f2v r;
    __builtin_memcpy(&r, p, sizeof(r));
    return r;
}

__global__ __launch_bounds__(256) void roi_align_avg_wmma(
    const float* __restrict__ features,   // (B, C, H, W) fp32
    const float* __restrict__ rois,       // (K, 5) fp32: b, x1, y1, x2, y2
    float* __restrict__ out)              // (K, C, 7, 7) fp32
{
    __shared__ int   s_y0[16];
    __shared__ float s_ly[16];
    __shared__ float s_vy[16];
    __shared__ int   s_x0[16];
    __shared__ float s_lx[16];
    __shared__ float s_vx[16];

    const int k = blockIdx.x;
    const int t = threadIdx.x;

    const float rb  = rois[k * 5 + 0];
    const float rx1 = rois[k * 5 + 1];
    const float ry1 = rois[k * 5 + 2];
    const float rx2 = rois[k * 5 + 3];
    const float ry2 = rois[k * 5 + 4];

    // Batch index is uniform across the block; force it scalar so the
    // feature base pointer lives in SGPRs and loads take the saddr form.
    const int b = __builtin_amdgcn_readfirstlane((int)rb);
    const float* __restrict__ fb = features + (size_t)b * (size_t)(C_DIM * HW_DIM);

    // ---- per-RoI tap tables (16 y samples, 16 x samples) ----
    if (t < 16) {
        const float y1f = ry1 * SCALE_F, y2f = ry2 * SCALE_F;
        const float bh  = fmaxf(y2f - y1f, 1.0f) * 0.125f;   // rh / PH
        const int ph = t >> 1, sr = t & 1;
        const float y  = y1f + ((float)ph + ((float)sr + 0.5f) * 0.5f) * bh;
        const float vy = (y > -1.0f && y < (float)H_DIM) ? 1.0f : 0.0f;
        const float yc = fminf(fmaxf(y, 0.0f), (float)(H_DIM - 1));
        const int   y0 = (int)yc;
        s_y0[t] = y0;
        s_ly[t] = yc - (float)y0;
        s_vy[t] = vy;
    } else if (t < 32) {
        const int sx = t - 16;
        const float x1f = rx1 * SCALE_F, x2f = rx2 * SCALE_F;
        const float bw  = fmaxf(x2f - x1f, 1.0f) * 0.125f;   // rw / PW
        const int pw = sx >> 1, sr = sx & 1;
        const float x  = x1f + ((float)pw + ((float)sr + 0.5f) * 0.5f) * bw;
        const float vx = (x > -1.0f && x < (float)W_DIM) ? 1.0f : 0.0f;
        const float xc = fminf(fmaxf(x, 0.0f), (float)(W_DIM - 1));
        const int   x0 = (int)xc;
        s_x0[sx] = x0;
        s_lx[sx] = xc - (float)x0;
        s_vx[sx] = vx;
    }
    __syncthreads();

    const int lane = t & 31;
    const int wave = t >> 5;
    const int g    = lane >> 4;    // K-half group for A/B operands
    const int m    = lane & 15;    // A row (channel-in-tile) == B/D column

    // This lane's 8 x-tap parameter sets (sx = 8*g + q), register-resident.
    unsigned xbr[8];
    float    lxr[8], vxr[8];
    #pragma unroll
    for (int q = 0; q < 8; ++q) {
        const int sx = 8 * g + q;
        const int x0 = s_x0[sx];
        xbr[q] = (unsigned)min(x0, W_DIM - 2);
        lxr[q] = (x0 >= W_DIM - 1) ? 1.0f : s_lx[sx];
        vxr[q] = s_vx[sx];
    }

    // Per-lane chunk-invariant B patterns for the 4 output tiles:
    // Bpat[nt][e] = 1/16 if (unsigned)(8*g + (e&7) - 2*j) < 4, for this lane's column
    // ij = nt*16 + m.  okm masks columns beyond the 49 real outputs.
    // Per-chunk activation: okc <=> (unsigned)(chunk - i) < 2  (g-independent).
    int iv4[4], okm4[4];
    v8i bp4[4];
    #pragma unroll
    for (int nt = 0; nt < 4; ++nt) {
        const int ij = nt * 16 + m;
        const int i  = ij / 7;
        const int j  = ij - 7 * i;
        iv4[nt]  = i;
        okm4[nt] = (ij < NOUT) ? -1 : 0;
        v16h bp = {};
        #pragma unroll
        for (int e = 0; e < 16; ++e) {
            bp[e] = ((unsigned)(8 * g + (e & 7) - 2 * j) < 4u) ? (_Float16)0.0625f
                                                 : (_Float16)0.0f;
        }
        __builtin_memcpy(&bp4[nt], &bp, sizeof(v8i));
    }

    // ---- channel tiles: 16 tiles of 16 channels, 8 waves round-robin ----
    for (int ctile = wave; ctile < C_DIM / 16; ctile += 8) {
        const int      c    = ctile * 16 + m;
        const unsigned cOff = (unsigned)c * (unsigned)HW_DIM;   // element offset

        v8f acc[4];
        acc[0] = (v8f){}; acc[1] = (v8f){}; acc[2] = (v8f){}; acc[3] = (v8f){};

        #pragma unroll 1   // keep one K-chunk in flight: <256 VGPRs, high occupancy
        for (int chunk = 0; chunk < 8; ++chunk) {
            // Prefetch next chunk's first feature row (at xbr[0] and xbr[7]) toward this WGP.
            if (chunk < 7) {
                const int      syn  = 2 * chunk + 2;
                const unsigned rowN = cOff + (unsigned)s_y0[syn] * (unsigned)W_DIM;
                __builtin_prefetch(fb + (rowN + xbr[0]), 0, 3);
                __builtin_prefetch(fb + (rowN + xbr[7]), 0, 3);
            }

            // Build A: this lane's 2x8 sample patch in f16 (A-layout exact).
            v16h A = {};
            #pragma unroll
            for (int syl = 0; syl < 2; ++syl) {
                const int      sy   = 2 * chunk + syl;
                const int      y0   = s_y0[sy];
                const float    ly   = s_ly[sy];
                const float    vy   = s_vy[sy];
                const int      y1   = min(y0 + 1, H_DIM - 1);
                const float    hy   = 1.0f - ly;
                const unsigned row0 = cOff + (unsigned)y0 * (unsigned)W_DIM;
                const unsigned row1 = cOff + (unsigned)y1 * (unsigned)W_DIM;
                #pragma unroll
                for (int q = 0; q < 8; ++q) {
                    const float lx = lxr[q], hx = 1.0f - lx;
                    const f2v p0 = ld_pair(fb + (row0 + xbr[q]));
                    const f2v p1 = ld_pair(fb + (row1 + xbr[q]));
                    const float val = (vy * vxr[q]) *
                        (hy * (hx * p0.x + lx * p0.y) +
                         ly * (hx * p1.x + lx * p1.y));
                    A[syl * 8 + q] = (_Float16)val;
                }
            }

            // Masked B operands, distinct staging per tile (no WAR chain):
            // B = (okc ? -1 : 0) & Bpat.
            v16h Bv[4];
            #pragma unroll
            for (int nt = 0; nt < 4; ++nt) {
                const int sel = okm4[nt] &
                    (((unsigned)(chunk - iv4[nt]) < 2u) ? -1 : 0);
                const v8i mskv = {sel, sel, sel, sel, sel, sel, sel, sel};
                const v8i bb = bp4[nt] & mskv;
                __builtin_memcpy(&Bv[nt], &bb, sizeof(v16h));
            }

            // Straight-line WMMA issue (no branches: keeps accumulators on a
            // single control path, avoids phi-copy churn; EXEC all-ones).
            #pragma unroll
            for (int nt = 0; nt < 4; ++nt) {
                acc[nt] = __builtin_amdgcn_wmma_f32_16x16x32_f16(
                    /*neg_a=*/false, A, /*neg_b=*/false, Bv[nt],
                    /*c_mod=*/(short)0, acc[nt],
                    /*reuse_a=*/false, /*reuse_b=*/false);
            }
        }

        // ---- store D (non-temporal: write-once stream, keep L2 for features)
        // lane m = column ij, VGPR r = channel r + 8*g.
        const size_t cbase = ((size_t)k * C_DIM + (size_t)(ctile * 16 + 8 * g));
        #pragma unroll
        for (int nt = 0; nt < 4; ++nt) {
            const int ij = nt * 16 + m;
            if (ij < NOUT) {
                #pragma unroll
                for (int r = 0; r < 8; ++r) {
                    __builtin_nontemporal_store(
                        acc[nt][r],
                        &out[(cbase + (size_t)r) * NOUT + (size_t)ij]);
                }
            }
        }
    }
}

extern "C" void kernel_launch(void* const* d_in, const int* in_sizes, int n_in,
                              void* d_out, int out_size, void* d_ws, size_t ws_size,
                              hipStream_t stream) {
    const float* features = (const float*)d_in[0];   // (4, 256, 200, 304) fp32
    const float* rois     = (const float*)d_in[1];   // (K, 5) fp32
    float* out = (float*)d_out;                      // (K, 256, 7, 7) fp32

    const int K = in_sizes[1] / 5;                   // 1024
    roi_align_avg_wmma<<<K, 256, 0, stream>>>(features, rois, out);
}
